// SPM_1735166787958
// MI455X (gfx1250) — matrix operand order repaired and checked by reading, not compiled, not run on hardware
//
#include <hip/hip_runtime.h>

// MI455X / gfx1250, wave32. Channel-attention (DANet CAM) pipeline:
//   E = Q Q^T (bf16 WMMA, f32 acc) -> softmax(rowmax-E) == exp(min-e)/sum -> O = A Q + x
// Tile feeds: GLOBAL_LOAD_ASYNC_TO_LDS_B128 (ASYNCcnt) + double-buffered LDS.
// Wave tile 64x32 (acc=64 VGPRs) -- keeps total ~130 VGPRs so codegen stays mov-free.
typedef __attribute__((ext_vector_type(16))) __bf16        v16bf;
typedef __attribute__((ext_vector_type(8)))  float         v8f;
typedef __attribute__((ext_vector_type(4)))  unsigned int  u32x4;
typedef __attribute__((ext_vector_type(2)))  unsigned int  u32x2;
typedef __attribute__((ext_vector_type(4)))  int           i32x4;

#define B_    16
#define C_    1024
#define N_    2304      // 48*48
#define TM    128       // block tile M
#define TN    128       // block tile N
#define KSTEP 32
#define LDSW  48        // padded LDS row (96B: 16B aligned, spreads banks)
#define AROW  2048      // attention bf16 row stride (== energy f32 row in bytes)

#define AS1 __attribute__((address_space(1)))
#define AS3 __attribute__((address_space(3)))

#if defined(__has_builtin)
#  if __has_builtin(__builtin_amdgcn_global_load_async_to_lds_b128)
#    define HAS_ASYNC_LDS 1
#  endif
#  if __has_builtin(__builtin_amdgcn_s_wait_asynccnt)
#    define HAS_WAIT_ASYNC 1
#  endif
#endif
#ifndef HAS_ASYNC_LDS
#  define HAS_ASYNC_LDS 0
#endif
#ifndef HAS_WAIT_ASYNC
#  define HAS_WAIT_ASYNC 0
#endif

__device__ __forceinline__ void async16(const unsigned short* g, unsigned short* l) {
#if HAS_ASYNC_LDS
    __builtin_amdgcn_global_load_async_to_lds_b128((AS1 i32x4*)(g), (AS3 i32x4*)(l), 0, 0);
#else
    *(u32x4*)l = *(const u32x4*)g;            // staged fallback (load+ds_store)
#endif
}

template <int N>
__device__ __forceinline__ void wait_async() {
#if HAS_WAIT_ASYNC
    __builtin_amdgcn_s_wait_asynccnt((unsigned short)N);
#else
    asm volatile("s_wait_asynccnt %0" ::"i"(N) : "memory");
#endif
}

__device__ __forceinline__ unsigned short f2bf(float f) {
    unsigned int u = __float_as_uint(f);
    u += 0x7FFFu + ((u >> 16) & 1u);          // RNE
    return (unsigned short)(u >> 16);
}

// ---------------- kernel 1: fp32 -> bf16 ----------------
__global__ __launch_bounds__(256) void cvt_bf16_kernel(const float* __restrict__ x,
                                                       unsigned short* __restrict__ q) {
    int i = blockIdx.x * blockDim.x + threadIdx.x;      // one float4 per thread
    float4 v = ((const float4*)x)[i];
    union { unsigned short s[4]; u32x2 u; } o;
    o.s[0] = f2bf(v.x); o.s[1] = f2bf(v.y); o.s[2] = f2bf(v.z); o.s[3] = f2bf(v.w);
    ((u32x2*)q)[i] = o.u;
}

// ---------------- kernel 2: Q^T (bf16, per batch [n][c]) ----------------
__global__ __launch_bounds__(256) void transpose_kernel(const unsigned short* __restrict__ q,
                                                        unsigned short* __restrict__ qt) {
    __shared__ unsigned short T[64][72];       // 144B row stride: 16B aligned
    const int bz = blockIdx.z;
    const int c0 = blockIdx.y * 64;
    const int n0 = blockIdx.x * 64;
    const unsigned short* qb = q  + (size_t)bz * C_ * N_;
    unsigned short*      qtb = qt + (size_t)bz * N_ * C_;
    const int tid = threadIdx.x;
    const int row = tid >> 2;                  // 0..63
    const int seg = (tid & 3) * 16;            // 0,16,32,48
    const u32x4* g = (const u32x4*)(qb + (size_t)(c0 + row) * N_ + n0 + seg);
    *(u32x4*)&T[row][seg]     = g[0];
    *(u32x4*)&T[row][seg + 8] = g[1];
    __syncthreads();
    union { unsigned short s[8]; u32x4 u; } o0, o1;
    #pragma unroll
    for (int j = 0; j < 8; ++j) { o0.s[j] = T[seg + j][row]; o1.s[j] = T[seg + 8 + j][row]; }
    u32x4* go = (u32x4*)(qtb + (size_t)(n0 + row) * C_ + c0 + seg);
    go[0] = o0.u; go[1] = o1.u;
}

// ---------------- WMMA fragment helpers ----------------
union Frag { v16bf v; u32x4 u[2]; };
// A frag (ISA 16-bit 16x32): lanes0-15 K{0..7,16..23}, lanes16-31 K{8..15,24..31}
__device__ __forceinline__ void load_afrag(Frag& f, const unsigned short (*As)[LDSW],
                                           int row, int kh) {
    f.u[0] = *(const u32x4*)&As[row][kh * 8];
    f.u[1] = *(const u32x4*)&As[row][kh * 8 + 16];
}
// B frag: lane n = column, lanes0-15 K{0..7,16..23}, lanes16-31 K{8..15,24..31}
__device__ __forceinline__ void load_bfrag(Frag& f, const unsigned short (*Bs)[LDSW],
                                           int col, int kh) {
    f.u[0] = *(const u32x4*)&Bs[col][kh * 8];
    f.u[1] = *(const u32x4*)&Bs[col][kh * 8 + 16];
}

// ---------------- double-buffered 128x128 WMMA GEMM body ----------------
// gA: row-major [TM rows, stride strideA], gB: row-major [TN rows, stride strideB]
// (for E=QQ^T, B rows are Q rows; for O=AQ, B rows are Q^T rows)
template <int NK>
__device__ __forceinline__ void gemm_body(const unsigned short* gA, int strideA,
                                          const unsigned short* gB, int strideB,
                                          unsigned short (*As)[TM][LDSW],
                                          unsigned short (*Bs)[TN][LDSW],
                                          v8f (&acc)[4][2]) {
    const int tid   = threadIdx.x;
    const int lane  = tid & 31;
    const int wid   = tid >> 5;
    const int waveM = wid & 1;                 // 2 x 4 wave grid, wave tile 64x32
    const int waveN = wid >> 1;
    const int r  = lane & 15;
    const int kh = lane >> 4;
    const int lrow = tid >> 1;                 // loader: 128 rows x 2 halves
    const int lseg = (tid & 1) * 16;

    // prologue: stage k-step 0 into buffer 0 (4 async b128 per thread)
    {
        const unsigned short* pa = gA + (size_t)lrow * strideA + lseg;
        async16(pa,     &As[0][lrow][lseg]);
        async16(pa + 8, &As[0][lrow][lseg + 8]);
        const unsigned short* pb = gB + (size_t)lrow * strideB + lseg;
        async16(pb,     &Bs[0][lrow][lseg]);
        async16(pb + 8, &Bs[0][lrow][lseg + 8]);
    }

    for (int ks = 0; ks < NK; ++ks) {
        const int cur = ks & 1;
        if (ks + 1 < NK) {
            const int k0n = (ks + 1) * KSTEP;
            const unsigned short* pa = gA + (size_t)lrow * strideA + k0n + lseg;
            async16(pa,     &As[cur ^ 1][lrow][lseg]);
            async16(pa + 8, &As[cur ^ 1][lrow][lseg + 8]);
            const unsigned short* pb = gB + (size_t)lrow * strideB + k0n + lseg;
            async16(pb,     &Bs[cur ^ 1][lrow][lseg]);
            async16(pb + 8, &Bs[cur ^ 1][lrow][lseg + 8]);
            wait_async<4>();                   // 4 newest (next tile) may stay in flight
        } else {
            wait_async<0>();
        }
        __syncthreads();                       // cur tile visible to all waves

        Frag a[4];
        #pragma unroll
        for (int mi = 0; mi < 4; ++mi)
            load_afrag(a[mi], As[cur], waveM * 64 + mi * 16 + r, kh);
        #pragma unroll
        for (int ni = 0; ni < 2; ++ni) {
            Frag bfr;
            load_bfrag(bfr, Bs[cur], waveN * 32 + ni * 16 + r, kh);
            #pragma unroll
            for (int mi = 0; mi < 4; ++mi)
                acc[mi][ni] = __builtin_amdgcn_wmma_f32_16x16x32_bf16(
                    false, a[mi].v, false, bfr.v, (short)0, acc[mi][ni], false, false);
        }
        __syncthreads();                       // reads of cur done before it is refilled
    }
}

// ---------------- kernel 3: E = Q Q^T ----------------
__global__ __launch_bounds__(256) void gemm_qqt_kernel(const unsigned short* __restrict__ q,
                                                       float* __restrict__ energy) {
    __shared__ unsigned short As[2][TM][LDSW];
    __shared__ unsigned short Bs[2][TN][LDSW];
    const int bz = blockIdx.z;
    const int M0 = blockIdx.y * TM;
    const int N0 = blockIdx.x * TN;
    const unsigned short* qb = q + (size_t)bz * C_ * N_;

    v8f acc[4][2] = {};
    gemm_body<N_ / KSTEP>(qb + (size_t)M0 * N_, N_, qb + (size_t)N0 * N_, N_, As, Bs, acc);

    float* eb = energy + (size_t)bz * C_ * C_;
    const int lane  = threadIdx.x & 31;
    const int wid   = threadIdx.x >> 5;
    const int waveM = wid & 1, waveN = wid >> 1;
    const int r = lane & 15, moff = (lane >> 4) * 8;
    #pragma unroll
    for (int mi = 0; mi < 4; ++mi)
        #pragma unroll
        for (int ni = 0; ni < 2; ++ni) {
            int col = N0 + waveN * 32 + ni * 16 + r;
            #pragma unroll
            for (int rr = 0; rr < 8; ++rr) {
                int row = M0 + waveM * 64 + mi * 16 + moff + rr;
                eb[(size_t)row * C_ + col] = acc[mi][ni][rr];
            }
        }
}

// ---------------- kernel 4: softmax(rowmax-E) == exp(min-e)/sum; bf16 in-place ----------------
__global__ __launch_bounds__(256) void softmax_kernel(float* __restrict__ energy) {
    const int row = blockIdx.x;                // 0 .. B_*C_-1
    float* e = energy + (size_t)row * C_;
    const int tid = threadIdx.x;
    float4 v = ((const float4*)e)[tid];

    float m = fminf(fminf(v.x, v.y), fminf(v.z, v.w));
    #pragma unroll
    for (int off = 16; off > 0; off >>= 1) m = fminf(m, __shfl_xor(m, off, 32));
    __shared__ float redm[8];
    if ((tid & 31) == 0) redm[tid >> 5] = m;
    __syncthreads();
    #pragma unroll
    for (int i = 0; i < 8; ++i) m = fminf(m, redm[i]);

    float t0 = __expf(m - v.x), t1 = __expf(m - v.y);
    float t2 = __expf(m - v.z), t3 = __expf(m - v.w);
    float s = t0 + t1 + t2 + t3;
    #pragma unroll
    for (int off = 16; off > 0; off >>= 1) s += __shfl_xor(s, off, 32);
    __shared__ float reds[8];
    if ((tid & 31) == 0) reds[tid >> 5] = s;
    __syncthreads();
    s = 0.f;
    #pragma unroll
    for (int i = 0; i < 8; ++i) s += reds[i];
    float inv = 1.0f / s;

    unsigned short* at = (unsigned short*)energy + (size_t)row * AROW;
    union { unsigned short h[4]; u32x2 u; } o;
    o.h[0] = f2bf(t0 * inv); o.h[1] = f2bf(t1 * inv);
    o.h[2] = f2bf(t2 * inv); o.h[3] = f2bf(t3 * inv);
    ((u32x2*)at)[tid] = o.u;                   // all row reads precede first barrier
}

// ---------------- kernel 5: O = A Q + x  (B rows come from Q^T) ----------------
__global__ __launch_bounds__(256) void gemm_av_kernel(const unsigned short* __restrict__ qt,
                                                      const unsigned short* __restrict__ attn,
                                                      const float* __restrict__ x,
                                                      float* __restrict__ out) {
    __shared__ unsigned short As[2][TM][LDSW];
    __shared__ unsigned short Bs[2][TN][LDSW];
    const int bz = blockIdx.z;
    const int M0 = blockIdx.y * TM;            // channel dim
    const int N0 = blockIdx.x * TN;            // spatial dim
    const unsigned short* ab  = attn + (size_t)bz * C_ * AROW;
    const unsigned short* qtb = qt   + (size_t)bz * N_ * C_;

    v8f acc[4][2] = {};
    gemm_body<C_ / KSTEP>(ab + (size_t)M0 * AROW, AROW,
                          qtb + (size_t)N0 * C_, C_, As, Bs, acc);

    const float* xb = x   + (size_t)bz * C_ * N_;
    float*       ob = out + (size_t)bz * C_ * N_;
    const int lane  = threadIdx.x & 31;
    const int wid   = threadIdx.x >> 5;
    const int waveM = wid & 1, waveN = wid >> 1;
    const int r = lane & 15, moff = (lane >> 4) * 8;
    #pragma unroll
    for (int mi = 0; mi < 4; ++mi)
        #pragma unroll
        for (int ni = 0; ni < 2; ++ni) {
            int col = N0 + waveN * 32 + ni * 16 + r;
            #pragma unroll
            for (int rr = 0; rr < 8; ++rr) {
                int row = M0 + waveM * 64 + mi * 16 + moff + rr;
                size_t idx = (size_t)row * N_ + col;
                ob[idx] = acc[mi][ni][rr] + xb[idx];   // fused residual
            }
        }
}

// ---------------- host launcher ----------------
extern "C" void kernel_launch(void* const* d_in, const int* in_sizes, int n_in,
                              void* d_out, int out_size, void* d_ws, size_t ws_size,
                              hipStream_t stream) {
    (void)in_sizes; (void)n_in; (void)out_size; (void)ws_size;
    const float* x = (const float*)d_in[0];
    const size_t qbytes = (size_t)B_ * C_ * N_ * sizeof(unsigned short);   // 75.5 MB
    unsigned short* qbf = (unsigned short*)d_ws;
    unsigned short* qtr = (unsigned short*)((char*)d_ws + qbytes);
    float* energy = (float*)((char*)d_ws + 2 * qbytes);                    // 64 MB
    float* out = (float*)d_out;

    const int total4 = (B_ * C_ * N_) / 4;
    cvt_bf16_kernel<<<total4 / 256, 256, 0, stream>>>(x, qbf);
    transpose_kernel<<<dim3(N_ / 64, C_ / 64, B_), 256, 0, stream>>>(qbf, qtr);
    gemm_qqt_kernel<<<dim3(C_ / TN, C_ / TM, B_), 256, 0, stream>>>(qbf, energy);
    softmax_kernel<<<B_ * C_, 256, 0, stream>>>(energy);
    gemm_av_kernel<<<dim3(N_ / TN, C_ / TM, B_), 256, 0, stream>>>(
        qtr, (const unsigned short*)energy, x, out);
}
